// Gtu_74672301408868
// MI455X (gfx1250) — matrix operand order repaired and checked by reading, not compiled, not run on hardware
//
#include <hip/hip_runtime.h>

// ---------------------------------------------------------------------------
// GTU (gated Toeplitz unit) for MI455X / gfx1250, wave32.
//   B=4  N=4096  E=512  H=8  D1=1536  D=192  R=512  L2=2N=8192
// GEMMs on v_wmma_f32_16x16x32_bf16 with double-buffered async global->LDS
// tile staging (ASYNCcnt). Toeplitz conv via LDS FFT (DIF fwd / conj-DIT inv,
// pointwise product in bit-reversed domain => no bit-reversal pass).
// ---------------------------------------------------------------------------

#define B_   4
#define N_   4096
#define E_   512
#define D1_  1536
#define R_   512
#define L2_  8192
#define PI_F 3.14159265358979323846f

typedef __bf16 bf16;
typedef __bf16        v16bf  __attribute__((ext_vector_type(16)));
typedef float         v8f    __attribute__((ext_vector_type(8)));
typedef unsigned int  u32x4  __attribute__((ext_vector_type(4)));
typedef float         f32x2  __attribute__((ext_vector_type(2)));

union FragU { u32x4 q[2]; v16bf v; };

#if defined(__gfx1250__) && __has_builtin(__builtin_amdgcn_global_load_async_to_lds_b128)
#define GTU_ASYNC 1
#if __has_builtin(__builtin_amdgcn_s_wait_asynccnt)
#define WAIT_ASYNC(n) __builtin_amdgcn_s_wait_asynccnt(n)
#else
#define WAIT_ASYNC(n) asm volatile("s_wait_asynccnt %0" ::"n"(n) : "memory")
#endif
// Builtin expects pointers to `int __vector(4)` in global (1) / LDS (3) spaces.
typedef int gv4i __attribute__((__vector_size__(16)));
#define GPTR(p) \
  ((__attribute__((address_space(1))) gv4i*)(__attribute__((address_space(1))) void*)(p))
#define LPTR(p) \
  ((__attribute__((address_space(3))) gv4i*)(__attribute__((address_space(3))) void*)(p))
#endif

// ------------------------------ elementwise --------------------------------
__global__ void k_cvt_bf16(const float* __restrict__ in, bf16* __restrict__ out, int n) {
  for (int i = blockIdx.x * blockDim.x + threadIdx.x; i < n; i += gridDim.x * blockDim.x)
    out[i] = (bf16)in[i];
}

// h0[n,r] = n * rpe_in_w[r] + rpe_in_b[r]   (position n; row 0 == position 0)
__global__ void k_rpe_input(const float* __restrict__ w, const float* __restrict__ b,
                            float* __restrict__ h0) {
  int i = blockIdx.x * blockDim.x + threadIdx.x;   // exactly N_*R_ threads
  int n = i >> 9, r = i & (R_ - 1);
  h0[i] = (float)n * w[r] + b[r];
}

// a[n,:] *= gamma^n  (causal decay; gamma^0 = 1)
__global__ void k_decay(float* __restrict__ a, int n) {
  int i = blockIdx.x * blockDim.x + threadIdx.x;
  if (i >= n) return;
  int row = i / D1_;
  a[i] *= __powf(0.99f, (float)row);
}

// SimpleRMSNorm + ReLU over rows of width 512, emit bf16 for next GEMM.
__global__ __launch_bounds__(256) void k_norm_relu(const float* __restrict__ in,
                                                   bf16* __restrict__ out) {
  const int row = blockIdx.x, tid = threadIdx.x;
  const float* p = in + (size_t)row * R_;
  float x0 = p[tid], x1 = p[tid + 256];
  float s = x0 * x0 + x1 * x1;
#pragma unroll
  for (int off = 16; off > 0; off >>= 1) s += __shfl_down(s, off, 32);
  __shared__ float part[8];
  if ((tid & 31) == 0) part[tid >> 5] = s;
  __syncthreads();
  float tot = 0.f;
#pragma unroll
  for (int i = 0; i < 8; ++i) tot += part[i];
  float rms = sqrtf(tot * (1.f / (float)R_));
  float inv = 1.f / (rms + 1e-8f);
  float y0 = x0 * inv; if (y0 < 0.f) y0 = 0.f;
  float y1 = x1 * inv; if (y1 < 0.f) y1 = 0.f;
  out[(size_t)row * R_ + tid]       = (bf16)y0;
  out[(size_t)row * R_ + tid + 256] = (bf16)y1;
}

// ------------------------------ transposes ---------------------------------
// weight (K x N) f32  ->  (N x K) bf16   (so B-fragments load contiguously)
__global__ __launch_bounds__(256) void k_wt_bf16(const float* __restrict__ in,
                                                 bf16* __restrict__ out, int K, int N) {
  __shared__ float t[32][33];
  int n0 = blockIdx.x * 32, k0 = blockIdx.y * 32;
  int tx = threadIdx.x, ty = threadIdx.y;
#pragma unroll
  for (int j = 0; j < 4; ++j)
    t[ty + 8 * j][tx] = in[(size_t)(k0 + ty + 8 * j) * N + n0 + tx];
  __syncthreads();
#pragma unroll
  for (int j = 0; j < 4; ++j)
    out[(size_t)(n0 + ty + 8 * j) * K + k0 + tx] = (bf16)t[tx][ty + 8 * j];
}

// (B*N, D1) f32 -> (B, D1, N) f32  : channel-contiguous layout for the FFT
__global__ __launch_bounds__(256) void k_transpose_bn(const float* __restrict__ in,
                                                      float* __restrict__ out) {
  __shared__ float t[32][33];
  int b = blockIdx.z, c0 = blockIdx.x * 32, n0 = blockIdx.y * 32;
  int tx = threadIdx.x, ty = threadIdx.y;
#pragma unroll
  for (int j = 0; j < 4; ++j)
    t[ty + 8 * j][tx] = in[((size_t)b * N_ + n0 + ty + 8 * j) * D1_ + c0 + tx];
  __syncthreads();
#pragma unroll
  for (int j = 0; j < 4; ++j)
    out[((size_t)b * D1_ + c0 + ty + 8 * j) * N_ + n0 + tx] = t[tx][ty + 8 * j];
}

// y[b,n,c] = bf16( u[b,n,c] * convT[b,c,n] )   (gate + transpose back)
__global__ __launch_bounds__(256) void k_gate_bf16(const float* __restrict__ u,
                                                   const float* __restrict__ convT,
                                                   bf16* __restrict__ y) {
  __shared__ float t[32][33];
  int b = blockIdx.z, c0 = blockIdx.x * 32, n0 = blockIdx.y * 32;
  int tx = threadIdx.x, ty = threadIdx.y;
#pragma unroll
  for (int j = 0; j < 4; ++j)
    t[ty + 8 * j][tx] = convT[((size_t)b * D1_ + c0 + ty + 8 * j) * N_ + n0 + tx];
  __syncthreads();
#pragma unroll
  for (int j = 0; j < 4; ++j) {
    size_t idx = ((size_t)b * N_ + n0 + ty + 8 * j) * D1_ + c0 + tx;
    y[idx] = (bf16)(u[idx] * t[tx][ty + 8 * j]);
  }
}

// ------------------------------ WMMA GEMM ----------------------------------
// C[M,Nc] = epi( A[M,K](bf16) @ Bt[Nc,K](bf16)^T + bias ),  f32 out.
// 128x128 block, 8 waves (2x4), wave = 4x2 tiles of 16x16, BK = 32.
// Double-buffered async global->LDS staging when available.
// EPI: 0 = bias, 1 = bias + silu.
template <int EPI>
__global__ __launch_bounds__(256) void k_gemm(const bf16* __restrict__ A,
                                              const bf16* __restrict__ Bt,
                                              const float* __restrict__ bias,
                                              float* __restrict__ C,
                                              int M, int Nc, int K) {
  __shared__ u32x4 As4[2][512];   // 2 x (128 rows x 32 bf16)
  __shared__ u32x4 Bs4[2][512];   // 2 x (128 cols x 32 bf16, rows of Bt)
  const int tid  = threadIdx.x;
  const int lane = tid & 31;
  const int wave = tid >> 5;
  const int wm = wave & 1;          // 2 waves -> 64 rows each
  const int wn = wave >> 1;         // 4 waves -> 32 cols each
  const int m0 = blockIdx.y * 128;
  const int n0 = blockIdx.x * 128;

  v8f acc[4][2];
#pragma unroll
  for (int a = 0; a < 4; ++a)
#pragma unroll
    for (int b = 0; b < 2; ++b)
#pragma unroll
      for (int r = 0; r < 8; ++r) acc[a][b][r] = 0.f;

  const int s0 = tid * 2;

  // fragment loads + 8 WMMAs on buffer `cur`
  auto compute = [&](int cur) {
    v16bf af[4], bfr[2];
    // A frag: row = lane%16; halves 0..7 -> K = 8*(lane/16)+0..7, 8..15 -> +16
#pragma unroll
    for (int tm = 0; tm < 4; ++tm) {
      int row = wm * 64 + tm * 16 + (lane & 15);
      int r4  = row * 4 + (lane >> 4);
      FragU u; u.q[0] = As4[cur][r4]; u.q[1] = As4[cur][r4 + 2];
      af[tm] = u.v;
    }
    // B frag: col = lane%16; halves 0..7 -> K = 8*(lane/16)+0..7, 8..15 -> +16
#pragma unroll
    for (int tn = 0; tn < 2; ++tn) {
      int col = wn * 32 + tn * 16 + (lane & 15);
      int b4  = col * 4 + (lane >> 4);
      FragU u; u.q[0] = Bs4[cur][b4]; u.q[1] = Bs4[cur][b4 + 2];
      bfr[tn] = u.v;
    }
#pragma unroll
    for (int tm = 0; tm < 4; ++tm)
#pragma unroll
      for (int tn = 0; tn < 2; ++tn)
        acc[tm][tn] = __builtin_amdgcn_wmma_f32_16x16x32_bf16(
            false, af[tm], false, bfr[tn], (short)0, acc[tm][tn], false, false);
  };

#ifdef GTU_ASYNC
  // ---- async double-buffered pipeline (GLOBAL_LOAD_ASYNC_TO_LDS_B128) ----
  auto stage = [&](int kb, int buf) {
#pragma unroll
    for (int t = 0; t < 2; ++t) {             // 4 async b128 per thread total
      int s = s0 + t;
      int row = s >> 2, ks = s & 3;
      __builtin_amdgcn_global_load_async_to_lds_b128(
          GPTR(A + (size_t)(m0 + row) * K + kb + ks * 8), LPTR(&As4[buf][s]), 0, 0);
      __builtin_amdgcn_global_load_async_to_lds_b128(
          GPTR(Bt + (size_t)(n0 + row) * K + kb + ks * 8), LPTR(&Bs4[buf][s]), 0, 0);
    }
  };
  const int nk = K >> 5;
  stage(0, 0);
  for (int i = 0; i < nk; ++i) {
    const int cur = i & 1;
    if (i + 1 < nk) {                         // prefetch next tile, then wait
      stage((i + 1) << 5, cur ^ 1);           //   for tile i only (in-order)
      WAIT_ASYNC(4);
    } else {
      WAIT_ASYNC(0);
    }
    __syncthreads();                          // tile i visible to all waves
    compute(cur);
    __syncthreads();                          // reads done before overwrite
  }
#else
  // ---- fallback: synchronous staging through VGPRs ----
  for (int kb = 0; kb < K; kb += 32) {
    __syncthreads();
#pragma unroll
    for (int t = 0; t < 2; ++t) {
      int s = s0 + t;
      int row = s >> 2, ks = s & 3;
      As4[0][s] = *(const u32x4*)(A  + (size_t)(m0 + row) * K + kb + ks * 8);
      Bs4[0][s] = *(const u32x4*)(Bt + (size_t)(n0 + row) * K + kb + ks * 8);
    }
    __syncthreads();
    compute(0);
  }
#endif

  // C/D: VGPR r -> M = r + 8*(lane>=16), N = lane%16
  const int rhi = 8 * (lane >> 4);
#pragma unroll
  for (int tm = 0; tm < 4; ++tm)
#pragma unroll
    for (int tn = 0; tn < 2; ++tn) {
      int col  = n0 + wn * 32 + tn * 16 + (lane & 15);
      float bi = bias[col];
#pragma unroll
      for (int r = 0; r < 8; ++r) {
        int row = m0 + wm * 64 + tm * 16 + rhi + r;
        float val = acc[tm][tn][r] + bi;
        if (EPI == 1) val = val * (1.f / (1.f + __expf(-val)));   // silu
        C[(size_t)row * Nc + col] = val;
      }
    }
}

// ------------------------------ FFT (LDS) ----------------------------------
// sgn < 0: forward DIF stage (h = L/2 .. 1), natural -> bit-reversed.
// sgn > 0: inverse DIT stage (h = 1 .. L/2), conj twiddles, bitrev -> natural.
__device__ __forceinline__ void fft_stage(float* re, float* im, int h, float sgn) {
  __syncthreads();
  for (int p = threadIdx.x; p < L2_ / 2; p += 256) {
    int j  = p & (h - 1);
    int i0 = 2 * p - j;
    int i1 = i0 + h;
    float ang = sgn * PI_F * (float)j / (float)h;
    float s, c;
    __sincosf(ang, &s, &c);
    float ar = re[i0], ai = im[i0], br = re[i1], bi = im[i1];
    if (sgn < 0.f) {                          // DIF butterfly
      re[i0] = ar + br; im[i0] = ai + bi;
      float dr = ar - br, di = ai - bi;
      re[i1] = dr * c - di * s;
      im[i1] = dr * s + di * c;
    } else {                                  // DIT butterfly
      float tr = br * c - bi * s;
      float ti = br * s + bi * c;
      re[i0] = ar + tr; im[i0] = ai + ti;
      re[i1] = ar - tr; im[i1] = ai - ti;
    }
  }
}

__device__ __forceinline__ void fft8192(float* re, float* im, bool inverse) {
  if (!inverse) { for (int h = L2_ / 2; h >= 1; h >>= 1) fft_stage(re, im, h, -1.f); }
  else          { for (int h = 1; h <= L2_ / 2; h <<= 1) fft_stage(re, im, h, +1.f); }
  __syncthreads();
}

// Forward FFT of Toeplitz coefficients, one (h,d) channel per workgroup.
__global__ __launch_bounds__(256) void k_fft_a(const float* __restrict__ a_raw,
                                               f32x2* __restrict__ ahat) {
  extern __shared__ float sm[];
  float* re = sm; float* im = sm + L2_;
  const int c = blockIdx.x;                       // 0 .. D1-1
  for (int i = threadIdx.x; i < L2_; i += 256) {
    re[i] = (i < N_) ? a_raw[(size_t)i * D1_ + c] : 0.f;
    im[i] = 0.f;
  }
  fft8192(re, im, false);
  for (int i = threadIdx.x; i < L2_; i += 256) {
    f32x2 t; t.x = re[i]; t.y = im[i];
    ahat[(size_t)c * L2_ + i] = t;
  }
}

// fwd FFT(v) * ahat, inverse FFT, scale 1/L, write conv back in place.
__global__ __launch_bounds__(256) void k_fft_conv(float* __restrict__ vT,
                                                  const f32x2* __restrict__ ahat) {
  extern __shared__ float sm[];
  float* re = sm; float* im = sm + L2_;
  const int bc = blockIdx.x;                      // b*D1 + c
  const int c  = bc % D1_;
  float* chan  = vT + (size_t)bc * N_;
  for (int i = threadIdx.x; i < L2_; i += 256) {
    re[i] = (i < N_) ? chan[i] : 0.f;
    im[i] = 0.f;
  }
  fft8192(re, im, false);
  const f32x2* ac = ahat + (size_t)c * L2_;
  for (int i = threadIdx.x; i < L2_; i += 256) {  // pointwise (bitrev domain)
    f32x2 w = ac[i];
    float r = re[i], m = im[i];
    re[i] = r * w.x - m * w.y;
    im[i] = r * w.y + m * w.x;
  }
  fft8192(re, im, true);
  const float sc = 1.f / (float)L2_;
  for (int i = threadIdx.x; i < N_; i += 256) chan[i] = re[i] * sc;
}

// ------------------------------ driver -------------------------------------
extern "C" void kernel_launch(void* const* d_in, const int* in_sizes, int n_in,
                              void* d_out, int out_size, void* d_ws, size_t ws_size,
                              hipStream_t stream) {
  const float* x         = (const float*)d_in[0];
  const float* u_w       = (const float*)d_in[1];
  const float* u_b       = (const float*)d_in[2];
  const float* v_w       = (const float*)d_in[3];
  const float* v_b       = (const float*)d_in[4];
  const float* o_w       = (const float*)d_in[5];
  const float* o_b       = (const float*)d_in[6];
  const float* rpe_in_w  = (const float*)d_in[7];
  const float* rpe_in_b  = (const float*)d_in[8];
  const float* rpe_ws    = (const float*)d_in[9];
  const float* rpe_bs    = (const float*)d_in[10];
  const float* rpe_out_w = (const float*)d_in[11];
  const float* rpe_out_b = (const float*)d_in[12];
  float* out = (float*)d_out;

  const int Mx = B_ * N_;  // 16384 rows
  char* cur = (char*)d_ws;
  auto alloc = [&](size_t bytes) -> void* {
    void* p = (void*)cur;
    cur += (bytes + 255) & ~(size_t)255;
    return p;
  };
  bf16*  xb      = (bf16*) alloc((size_t)Mx * E_ * 2);
  bf16*  u_wT    = (bf16*) alloc((size_t)D1_ * E_ * 2);
  bf16*  v_wT    = (bf16*) alloc((size_t)D1_ * E_ * 2);
  bf16*  o_wT    = (bf16*) alloc((size_t)E_ * D1_ * 2);
  bf16*  rpe_wT  = (bf16*) alloc((size_t)3 * R_ * R_ * 2);
  bf16*  rpeo_wT = (bf16*) alloc((size_t)D1_ * R_ * 2);
  float* u_f     = (float*)alloc((size_t)Mx * D1_ * 4);
  float* v_f     = (float*)alloc((size_t)Mx * D1_ * 4);
  float* vT      = (float*)alloc((size_t)Mx * D1_ * 4);   // then conv result
  float* h0      = (float*)alloc((size_t)N_ * R_ * 4);
  float* h1      = (float*)alloc((size_t)N_ * R_ * 4);
  bf16*  hact    = (bf16*) alloc((size_t)N_ * R_ * 2);
  float* a_raw   = (float*)alloc((size_t)N_ * D1_ * 4);
  f32x2* ahat    = (f32x2*)alloc((size_t)D1_ * L2_ * 8);
  bf16*  yb      = (bf16*) alloc((size_t)Mx * D1_ * 2);

  dim3 b256(256), b32x8(32, 8);

  // precision conversion + weight transposes (bf16)
  k_cvt_bf16<<<8192, b256, 0, stream>>>(x, xb, Mx * E_);
  k_wt_bf16<<<dim3(D1_ / 32, E_ / 32), b32x8, 0, stream>>>(u_w, u_wT, E_, D1_);
  k_wt_bf16<<<dim3(D1_ / 32, E_ / 32), b32x8, 0, stream>>>(v_w, v_wT, E_, D1_);
  k_wt_bf16<<<dim3(E_ / 32, D1_ / 32), b32x8, 0, stream>>>(o_w, o_wT, D1_, E_);
  for (int i = 0; i < 3; ++i)
    k_wt_bf16<<<dim3(R_ / 32, R_ / 32), b32x8, 0, stream>>>(
        rpe_ws + (size_t)i * R_ * R_, rpe_wT + (size_t)i * R_ * R_, R_, R_);
  k_wt_bf16<<<dim3(D1_ / 32, R_ / 32), b32x8, 0, stream>>>(rpe_out_w, rpeo_wT, R_, D1_);

  // gated projections (WMMA, silu epilogue)
  k_gemm<1><<<dim3(D1_ / 128, Mx / 128), b256, 0, stream>>>(xb, u_wT, u_b, u_f, Mx, D1_, E_);
  k_gemm<1><<<dim3(D1_ / 128, Mx / 128), b256, 0, stream>>>(xb, v_wT, v_b, v_f, Mx, D1_, E_);
  k_transpose_bn<<<dim3(D1_ / 32, N_ / 32, B_), b32x8, 0, stream>>>(v_f, vT);

  // RPE MLP (WMMA) -> Toeplitz coefficients
  k_rpe_input<<<(N_ * R_) / 256, b256, 0, stream>>>(rpe_in_w, rpe_in_b, h0);
  float* hin = h0; float* hout = h1;
  for (int i = 0; i < 3; ++i) {
    k_norm_relu<<<N_, b256, 0, stream>>>(hin, hact);
    k_gemm<0><<<dim3(R_ / 128, N_ / 128), b256, 0, stream>>>(
        hact, rpe_wT + (size_t)i * R_ * R_, rpe_bs + (size_t)i * R_, hout, N_, R_, R_);
    float* t = hin; hin = hout; hout = t;
  }
  k_norm_relu<<<N_, b256, 0, stream>>>(hin, hact);
  k_gemm<0><<<dim3(D1_ / 128, N_ / 128), b256, 0, stream>>>(
      hact, rpeo_wT, rpe_out_b, a_raw, N_, D1_, R_);
  k_decay<<<(N_ * D1_) / 256, b256, 0, stream>>>(a_raw, N_ * D1_);

  // Toeplitz multiply via length-2N FFT circular conv (64 KB LDS / WG)
  k_fft_a<<<D1_, b256, 2 * L2_ * sizeof(float), stream>>>(a_raw, ahat);
  k_fft_conv<<<B_ * D1_, b256, 2 * L2_ * sizeof(float), stream>>>(vT, ahat);

  // gate + output projection (WMMA)
  k_gate_bf16<<<dim3(D1_ / 32, N_ / 32, B_), b32x8, 0, stream>>>(u_f, vT, yb);
  k_gemm<0><<<dim3(E_ / 128, Mx / 128), b256, 0, stream>>>(yb, o_wT, o_b, out, Mx, E_, D1_);
}
